// AnisotropyFrame_79645873537099
// MI455X (gfx1250) — matrix operand order repaired and checked by reading, not compiled, not run on hardware
//
#include <hip/hip_runtime.h>
#include <math.h>

typedef __attribute__((ext_vector_type(16))) _Float16 v16h;
typedef __attribute__((ext_vector_type(8)))  float    v8f;

#define HIDDIM 128
#define MID    64
#define EPSV   1e-6f

// ---------------------------------------------------------------------------
// Kernel 1: pre[v][n] = h[v] @ W1[0:128,n] + curv[v] @ W1[128:132,n] + b1[n]
// One wave32 per block, 16 rows x 64 cols per wave via 4 WMMA accumulators,
// K padded 132 -> 160 (5 chunks of 32), f16 inputs / f32 accumulate.
// VGPR layouts per CDNA5 ISA 7.12.2:
//   A (16-bit, 16x32): lane L (g=L/16, m=L%16), element i -> K = (i/8)*16 + g*8 + i%8
//   B (16-bit, 32x16): lane L (g=L/16, n=L%16), element i -> K = (i/8)*16 + g*8 + i%8
//   C/D (f32 16x16):   VGPR r: lanes 0-15 -> M=r, N=lane; lanes 16-31 -> M=r+8
// ---------------------------------------------------------------------------
__global__ __launch_bounds__(32) void
pre_gemm_kernel(const float* __restrict__ h, const float* __restrict__ curv,
                const float* __restrict__ W1, const float* __restrict__ b1,
                float* __restrict__ pre, int V)
{
    const int lane = threadIdx.x & 31;
    const int g    = lane >> 4;   // half-wave group
    const int lm   = lane & 15;   // M for A, N-in-tile for B/C/D
    const int v0   = blockIdx.x << 4;

    v8f acc[4];
#pragma unroll
    for (int nt = 0; nt < 4; ++nt) {
        const float bv = b1[nt * 16 + lm];
#pragma unroll
        for (int r = 0; r < 8; ++r) acc[nt][r] = bv;
    }

    int row = v0 + lm;
    if (row >= V) row = V - 1;                 // clamp (V=50000 is exact anyway)
    const float* __restrict__ hrow = h + (size_t)row * HIDDIM;

#pragma unroll
    for (int kb = 0; kb < 5; ++kb) {
        // ---- A fragment ----
        v16h a;
        if (kb < 4) {  // K in [kb*32, kb*32+32) all come from h (K<128)
            const float* hp = hrow + kb * 32 + g * 8;
            const float4 f0 = *(const float4*)(hp + 0);
            const float4 f1 = *(const float4*)(hp + 4);
            const float4 f2 = *(const float4*)(hp + 16);
            const float4 f3 = *(const float4*)(hp + 20);
            a[0]=(_Float16)f0.x;  a[1]=(_Float16)f0.y;  a[2]=(_Float16)f0.z;  a[3]=(_Float16)f0.w;
            a[4]=(_Float16)f1.x;  a[5]=(_Float16)f1.y;  a[6]=(_Float16)f1.z;  a[7]=(_Float16)f1.w;
            a[8]=(_Float16)f2.x;  a[9]=(_Float16)f2.y;  a[10]=(_Float16)f2.z; a[11]=(_Float16)f2.w;
            a[12]=(_Float16)f3.x; a[13]=(_Float16)f3.y; a[14]=(_Float16)f3.z; a[15]=(_Float16)f3.w;
        } else {       // K in [128,160): 128..131 from curvature, rest zero pad
#pragma unroll
            for (int i = 0; i < 16; ++i) {
                const int K = 128 + ((i >> 3) << 4) + (g << 3) + (i & 7);
                const float val = (K < 132) ? curv[(size_t)row * 4 + (K - 128)] : 0.0f;
                a[i] = (_Float16)val;
            }
        }
        // ---- 4 N-tiles ----
#pragma unroll
        for (int nt = 0; nt < 4; ++nt) {
            const int n = nt * 16 + lm;
            v16h b;
#pragma unroll
            for (int i = 0; i < 16; ++i) {
                const int K = kb * 32 + ((i >> 3) << 4) + (g << 3) + (i & 7);
                const float w = (K < 132) ? W1[(size_t)K * MID + n] : 0.0f;
                b[i] = (_Float16)w;
            }
            acc[nt] = __builtin_amdgcn_wmma_f32_16x16x32_f16(
                false, a, false, b, (short)0, acc[nt], false, false);
        }
    }

    // ---- store D: VGPR r holds row m = g*8 + r, col n = nt*16 + lm ----
#pragma unroll
    for (int nt = 0; nt < 4; ++nt) {
#pragma unroll
        for (int r = 0; r < 8; ++r) {
            const int m = (g << 3) + r;
            if (v0 + m < V)
                pre[(size_t)(v0 + m) * MID + nt * 16 + lm] = acc[nt][r];
        }
    }
}

// ---------------------------------------------------------------------------
__global__ void zero_kernel(float* __restrict__ p, int n)
{
    const int i = blockIdx.x * blockDim.x + threadIdx.x;
    if (i < n) p[i] = 0.0f;
}

// ---------------------------------------------------------------------------
// Kernel 2: per-edge tail.  hmid = silu(pre[row] + dist * W1[132,:]),
// alpha = hmid @ W2 + b2, atomically scatter alpha * diff/dist into d[row].
// pre-gather is 256B/edge out of the 192MB L2 (pre is only 12.8MB resident).
// ---------------------------------------------------------------------------
__global__ __launch_bounds__(256) void
edge_kernel(const float* __restrict__ x, const int* __restrict__ ei,
            const float* __restrict__ pre, const float* __restrict__ W1,
            const float* __restrict__ W2, const float* __restrict__ b2,
            float* __restrict__ d, int E)
{
    __shared__ float s_w1d[MID];
    __shared__ float s_w2[MID];
    if (threadIdx.x < MID) {
        s_w1d[threadIdx.x] = W1[132 * MID + threadIdx.x];
        s_w2[threadIdx.x]  = W2[threadIdx.x];
    }
    __syncthreads();

    const int e = blockIdx.x * blockDim.x + threadIdx.x;
    if (e >= E) return;

    const int r = ei[e];
    const int c = ei[E + e];

    const float dx = x[3 * c + 0] - x[3 * r + 0];
    const float dy = x[3 * c + 1] - x[3 * r + 1];
    const float dz = x[3 * c + 2] - x[3 * r + 2];
    float dist = sqrtf(dx * dx + dy * dy + dz * dz);
    dist = fmaxf(dist, EPSV);

    const float* __restrict__ p = pre + (size_t)r * MID;
    float alpha = b2[0];
#pragma unroll 4
    for (int n = 0; n < MID; n += 4) {
        const float4 pv = *(const float4*)(p + n);
        const float z0 = fmaf(dist, s_w1d[n + 0], pv.x);
        const float z1 = fmaf(dist, s_w1d[n + 1], pv.y);
        const float z2 = fmaf(dist, s_w1d[n + 2], pv.z);
        const float z3 = fmaf(dist, s_w1d[n + 3], pv.w);
        alpha = fmaf(z0 * __frcp_rn(1.0f + __expf(-z0)), s_w2[n + 0], alpha);
        alpha = fmaf(z1 * __frcp_rn(1.0f + __expf(-z1)), s_w2[n + 1], alpha);
        alpha = fmaf(z2 * __frcp_rn(1.0f + __expf(-z2)), s_w2[n + 2], alpha);
        alpha = fmaf(z3 * __frcp_rn(1.0f + __expf(-z3)), s_w2[n + 3], alpha);
    }

    const float s = alpha / dist;   // alpha * unit scaling
    atomicAdd(&d[3 * r + 0], s * dx);
    atomicAdd(&d[3 * r + 1], s * dy);
    atomicAdd(&d[3 * r + 2], s * dz);
}

// ---------------------------------------------------------------------------
__global__ __launch_bounds__(256) void
norm_kernel(const float* __restrict__ d, float* __restrict__ out, int V)
{
    const int v = blockIdx.x * blockDim.x + threadIdx.x;
    if (v >= V) return;
    const float a = d[3 * v + 0], b = d[3 * v + 1], c = d[3 * v + 2];
    const float n = fmaxf(sqrtf(a * a + b * b + c * c), EPSV);
    const float inv = 1.0f / n;
    out[3 * v + 0] = a * inv;
    out[3 * v + 1] = b * inv;
    out[3 * v + 2] = c * inv;
}

// ---------------------------------------------------------------------------
extern "C" void kernel_launch(void* const* d_in, const int* in_sizes, int n_in,
                              void* d_out, int out_size, void* d_ws, size_t ws_size,
                              hipStream_t stream)
{
    const float* h    = (const float*)d_in[0];
    const float* x    = (const float*)d_in[1];
    const float* curv = (const float*)d_in[2];
    const int*   ei   = (const int*)d_in[3];
    const float* W1   = (const float*)d_in[4];
    const float* b1   = (const float*)d_in[5];
    const float* W2   = (const float*)d_in[6];
    const float* b2   = (const float*)d_in[7];
    float* out = (float*)d_out;

    const int V = in_sizes[1] / 3;
    const int E = in_sizes[3] / 2;

    float* pre  = (float*)d_ws;                 // V * 64 f32  (12.8 MB)
    float* dacc = pre + (size_t)V * MID;        // V * 3  f32  (0.6 MB)

    const int nAcc = V * 3;
    zero_kernel<<<(nAcc + 255) / 256, 256, 0, stream>>>(dacc, nAcc);

    pre_gemm_kernel<<<(V + 15) / 16, 32, 0, stream>>>(h, curv, W1, b1, pre, V);

    edge_kernel<<<(E + 255) / 256, 256, 0, stream>>>(x, ei, pre, W1, W2, b2, dacc, E);

    norm_kernel<<<(V + 255) / 256, 256, 0, stream>>>(dacc, out, V);
}
